// GPNN_VCOCO_59227599012625
// MI455X (gfx1250) — matrix operand order repaired and checked by reading, not compiled, not run on hardware
//
#include <hip/hip_runtime.h>
#include <hip/hip_bf16.h>
#include <math.h>

// ---------------------------------------------------------------------------
// GPNN-VCOCO pipeline for gfx1250 (MI455X), bf16 WMMA core + TDM weight
// staging into LDS. B=32, N=64, CE=CN=M=LH=256, L=3, HOI=26, ROLES=3.
// Rows R = B*N*N = 131072. All heavy GEMMs are R x 256 x 256.
// ---------------------------------------------------------------------------

#define RWS   131072          // B*N*N rows
#define NELEM 33554432        // RWS*256
#define BN    2048            // B*N

typedef __attribute__((ext_vector_type(16))) __bf16 v16bf;
typedef __attribute__((ext_vector_type(8)))  float  v8f;
typedef __attribute__((ext_vector_type(4)))  unsigned int v4u;
typedef __attribute__((ext_vector_type(4)))  unsigned int u32x4;
typedef __attribute__((ext_vector_type(8)))  int i32x8;
typedef __attribute__((ext_vector_type(4)))  int i32x4;

union bfvec { v16bf v; v4u q[2]; };

// LDS weight tile: 64 W rows, each 256 bf16 + 8 bf16 (16B) pad -> 528B row
// stride. Lane n (column) starts at bank (132*n)%64 = 4n -> conflict-free
// ds_load_b128 across the 16-lane column group.
#define WROWS   64
#define WPITCH  264            // ushorts per padded row (264*2 = 528 B)

#if defined(__has_builtin)
#if __has_builtin(__builtin_amdgcn_tensor_load_to_lds) && \
    __has_builtin(__builtin_amdgcn_s_wait_tensorcnt)
#define USE_TDM 1
#endif
#endif
#ifndef USE_TDM
#define USE_TDM 0
#endif

__device__ __forceinline__ unsigned short f32_to_bf16(float f) {
  unsigned int u = __float_as_uint(f);
  u += 0x7FFFu + ((u >> 16) & 1u);          // round-to-nearest-even
  return (unsigned short)(u >> 16);
}
__device__ __forceinline__ float bf16_to_f32(unsigned short h) {
  return __uint_as_float(((unsigned int)h) << 16);
}
__device__ __forceinline__ float sigmoidf(float x) {
  return 1.0f / (1.0f + __expf(-x));
}

#if USE_TDM
// Tensor DMA: copy 64 rows x 256 bf16 of W (row stride 256 elems) into LDS,
// inserting 4 DWORDs (16B) of pad every 128 DWORDs (512B = one row).
// D# per CDNA5 ISA 8.3/8.4: group0 = {flags, lds_addr, global_addr, type=2},
// group1 = {ds/pad flags, tensor_dim0/1, tile_dim0/1/2, strides}.
// clang-23 therock builtin takes 6 args: (g0, g1, g2, g3, extra, cpol).
__device__ __forceinline__ void tdm_stage_w(const unsigned short* gsrc,
                                            unsigned int lds_off) {
  unsigned long long ga = (unsigned long long)(size_t)gsrc;
  u32x4 g0;
  g0[0] = 1u;                                        // count=1, user mode
  g0[1] = lds_off;                                   // lds_addr (bytes)
  g0[2] = (unsigned int)(ga & 0xFFFFFFFFu);          // global_addr[31:0]
  g0[3] = (unsigned int)((ga >> 32) & 0x01FFFFFFu)   // global_addr[56:32]
          | 0x80000000u;                             // type=2 ("image")
  i32x8 g1;
  g1[0] = (int)((1u << 16)      // data_size = 1 -> 2-byte elements
              | (1u << 20)      // pad_enable
              | (6u << 22)      // pad_interval = 128 DWORDs (one 512B row)
              | (3u << 25));    // pad_amount   = 4 DWORDs (16B)
  g1[1] = (int)(256u << 16);    // tensor_dim0 = 256 (bits 63:48)
  g1[2] = (int)(64u  << 16);    // tensor_dim1 = 64  (bits 95:80)
  g1[3] = (int)(256u << 16);    // tile_dim0   = 256 (bits 127:112)
  g1[4] = 64;                   // tile_dim1   = 64  (bits 143:128)
  g1[5] = 256;                  // tensor_dim0_stride = 256 (bits 191:160)
  g1[6] = 0;
  g1[7] = 0;
  i32x4 z4 = {0, 0, 0, 0};
  i32x8 z8 = {0, 0, 0, 0, 0, 0, 0, 0};
  __builtin_amdgcn_tensor_load_to_lds(g0, g1, z4, z4, z8, 0);
}
#endif

enum { MODE_BASE = 0, MODE_G = 1, MODE_L1 = 2, MODE_L2 = 3 };

// ---------------------------------------------------------------------------
// Core GEMM: Y[p,o] = sum_k X[p,k]*W[o,k], bf16 in, f32 accum via
// v_wmma_f32_16x16x32_bf16. One wave owns a 16-row strip; the block stages W
// into LDS in 64-row quarters (TDM) so B operands come from ds_load_b128
// and are reused by all 8 waves.
//   MODE_BASE: Y = relu(D + mb[o] + hb[(b,w),o])          -> bf16
//   MODE_G   : Y = D                                       -> bf16
//   MODE_L1  : Y = relu(D + lb1[o])                        -> bf16
//   MODE_L2  : H2 = relu(D + lb2[o]); A = H2 . lw3 + lb3;  -> A f32, S f32
// ---------------------------------------------------------------------------
template <int MODE>
__global__ __launch_bounds__(256) void gemm_bf16_wmma(
    const unsigned short* __restrict__ X,     // [RWS,256] bf16 bits
    const unsigned short* __restrict__ W,     // [256,256] bf16 bits, (out,in)
    unsigned short* __restrict__ Ybf,         // bf16 out (BASE/G/L1)
    const float* __restrict__ bias,           // per-col bias (mb/lb1/lb2)
    const float* __restrict__ hb,             // MODE_BASE: [BN,256]
    const float* __restrict__ lw3,            // MODE_L2: [256]
    const float* __restrict__ lb3p,           // MODE_L2: [1]
    float* __restrict__ Aout,                 // MODE_L2: [RWS]
    float* __restrict__ Sout)                 // MODE_L2: [RWS]
{
  __shared__ __align__(16) unsigned short wlds[WROWS * WPITCH];  // 33,792 B

  const int wave = threadIdx.x >> 5;
  const int lane = threadIdx.x & 31;
  const int g    = lane >> 4;                 // half-wave group
  const int r    = lane & 15;                 // A row idx / B-D col idx
  const int row_base = (blockIdx.x * 8 + wave) * 16;

  // ---- A operand: full K=256 of this lane's row, kept in VGPRs (64 VGPRs)
  const unsigned short* xrow = X + (size_t)(row_base + r) * 256;
  v16bf a[8];
#pragma unroll
  for (int kk = 0; kk < 8; ++kk) {
    bfvec t;
    t.q[0] = *(const v4u*)(xrow + kk * 32 + g * 8);        // K = kk*32+g*8 ..+7
    t.q[1] = *(const v4u*)(xrow + kk * 32 + 16 + g * 8);   // K = +16
    a[kk] = t.v;
  }

  float acc[8];
  if (MODE == MODE_L2) {
#pragma unroll
    for (int i = 0; i < 8; ++i) acc[i] = 0.0f;
  }

  for (int q = 0; q < 4; ++q) {
    __syncthreads();  // protect previous quarter's LDS reads before overwrite

#if USE_TDM
    if (threadIdx.x == 0)
      tdm_stage_w(W + (size_t)q * WROWS * 256, (unsigned int)(size_t)wlds);
    __builtin_amdgcn_s_wait_tensorcnt(0);   // no-op for non-issuing waves
#else
    {
      const v4u* src = (const v4u*)(W + (size_t)q * WROWS * 256);
      v4u* dst = (v4u*)wlds;                // 33 v4u per padded row
      for (int i = threadIdx.x; i < WROWS * 32; i += 256)
        dst[(i >> 5) * 33 + (i & 31)] = src[i];
    }
#endif
    if (q < 3 && threadIdx.x < WROWS)       // warm next quarter in L2
      __builtin_prefetch(W + (size_t)(q + 1) * WROWS * 256 + threadIdx.x * 256,
                         0, 1);
    __syncthreads();

    for (int ct2 = 0; ct2 < 4; ++ct2) {
      const int o = q * WROWS + ct2 * 16 + r;            // output channel
      const unsigned short* wrow = wlds + (ct2 * 16 + r) * WPITCH;

      v8f c = {};
#pragma unroll
      for (int kk = 0; kk < 8; ++kk) {
        bfvec bv;
        bv.q[0] = *(const v4u*)(wrow + kk * 32 + g * 8);       // K = g*8+0..7
        bv.q[1] = *(const v4u*)(wrow + kk * 32 + 16 + g * 8);  // K = 16+g*8+0..7
        c = __builtin_amdgcn_wmma_f32_16x16x32_bf16(
                false, a[kk], false, bv.v, (short)0, c, false, false);
      }

      // epilogue: lane holds D rows (g*8+i), column o
      if (MODE == MODE_G) {
#pragma unroll
        for (int i = 0; i < 8; ++i)
          Ybf[(size_t)(row_base + g * 8 + i) * 256 + o] = f32_to_bf16(c[i]);
      } else if (MODE == MODE_L1) {
        const float bo = bias[o];
#pragma unroll
        for (int i = 0; i < 8; ++i) {
          float y = c[i] + bo;
          y = y > 0.0f ? y : 0.0f;
          Ybf[(size_t)(row_base + g * 8 + i) * 256 + o] = f32_to_bf16(y);
        }
      } else if (MODE == MODE_BASE) {
        const float bo = bias[o];
#pragma unroll
        for (int i = 0; i < 8; ++i) {
          const int p     = row_base + g * 8 + i;            // b*4096+v*64+w
          const int hbrow = ((p >> 12) << 6) | (p & 63);     // b*64+w
          float y = c[i] + bo + hb[(size_t)hbrow * 256 + o];
          y = y > 0.0f ? y : 0.0f;
          Ybf[(size_t)p * 256 + o] = f32_to_bf16(y);
        }
      } else { // MODE_L2: fold layer-3 dot into the epilogue
        const float bo = bias[o];
        const float w3 = lw3[o];
#pragma unroll
        for (int i = 0; i < 8; ++i) {
          float y = c[i] + bo;
          y = y > 0.0f ? y : 0.0f;
          acc[i] += y * w3;
        }
      }
    }
  }

  if (MODE == MODE_L2) {
    const float lb3 = lb3p[0];
#pragma unroll
    for (int i = 0; i < 8; ++i) {
      float s = acc[i];
      s += __shfl_xor(s, 1, 32);
      s += __shfl_xor(s, 2, 32);
      s += __shfl_xor(s, 4, 32);
      s += __shfl_xor(s, 8, 32);               // reduce 16-lane column group
      if (r == 0) {
        const int p = row_base + g * 8 + i;
        const float av = s + lb3;
        Aout[p] = av;
        Sout[p] = sigmoidf(av);
      }
    }
  }
}

// ---------------------------------------------------------------------------
// Small helper kernels
// ---------------------------------------------------------------------------
__global__ void f2bf_kernel(const float* __restrict__ src,
                            unsigned short* __restrict__ dst, int n) {
  int i = blockIdx.x * blockDim.x + threadIdx.x;
  if (i < n) dst[i] = f32_to_bf16(src[i]);
}

// mw_e = mw[:, :256]  (row stride 512)
__global__ void f2bf_strided_kernel(const float* __restrict__ src,
                                    unsigned short* __restrict__ dst,
                                    int cols, int stride, int off) {
  int i = blockIdx.x * blockDim.x + threadIdx.x;
  int row = i / cols, col = i - row * cols;
  dst[i] = f32_to_bf16(src[(size_t)row * stride + off + col]);
}

// hb[(b,w), m] = sum_c node_features[b,w,c] * mw[m, 256+c]
__global__ __launch_bounds__(256) void hbase_kernel(
    const float* __restrict__ nf, const float* __restrict__ mw,
    float* __restrict__ hb) {
  __shared__ float row[256];
  const int bn = blockIdx.x, m = threadIdx.x;
  row[m] = nf[(size_t)bn * 256 + m];
  __syncthreads();
  const float* wr = mw + (size_t)m * 512 + 256;
  float s = 0.0f;
#pragma unroll 8
  for (int c = 0; c < 256; ++c) s += row[c] * wr[c];
  hb[(size_t)bn * 256 + m] = s;
}

// H1[row (b,v,w), c] = relu(S[b,w,v] * G[(b,w,v), c] + lb1[c])   (bf16 out)
__global__ __launch_bounds__(256) void scale_relu_kernel(
    const unsigned short* __restrict__ G, const float* __restrict__ S,
    const float* __restrict__ lb1, unsigned short* __restrict__ H1) {
  const int p = blockIdx.x, c = threadIdx.x;
  const int b = p >> 12, v = (p >> 6) & 63, w = p & 63;
  const int q = (b << 12) | (w << 6) | v;          // transposed row
  const float s = S[q];
  float y = s * bf16_to_f32(G[(size_t)q * 256 + c]) + lb1[c];
  H1[(size_t)p * 256 + c] = f32_to_bf16(y > 0.0f ? y : 0.0f);
}

// x[b,v,c] = sum_w S3[b,v,w] * base[(b,v,w), c]
__global__ __launch_bounds__(256) void xsum_kernel(
    const float* __restrict__ S, const unsigned short* __restrict__ baseb,
    float* __restrict__ x) {
  __shared__ float srow[64];
  const int bv = blockIdx.x, c = threadIdx.x;
  if (c < 64) srow[c] = S[(size_t)bv * 64 + c];
  __syncthreads();
  const unsigned short* bp = baseb + (size_t)bv * 64 * 256 + c;
  float s = 0.0f;
#pragma unroll 8
  for (int w = 0; w < 64; ++w) s += srow[w] * bf16_to_f32(bp[(size_t)w * 256]);
  x[(size_t)bv * 256 + c] = s;
}

// GRU cell + HOI / role heads, one block per (b,n)
__global__ __launch_bounds__(256) void gru_head_kernel(
    const float* __restrict__ x, const float* __restrict__ nf,
    const float* __restrict__ w_ih, const float* __restrict__ w_hh,
    const float* __restrict__ b_ih, const float* __restrict__ b_hh,
    const float* __restrict__ rw, const float* __restrict__ rb,
    const float* __restrict__ rw2, const float* __restrict__ rb2,
    float* __restrict__ out_labels, float* __restrict__ out_roles) {
  __shared__ float xr[256], hr[256], hn[256];
  const int bn = blockIdx.x, t = threadIdx.x;
  xr[t] = x[(size_t)bn * 256 + t];
  hr[t] = nf[(size_t)bn * 256 + t];
  __syncthreads();
  float gi[3], gh[3];
#pragma unroll
  for (int j = 0; j < 3; ++j) {
    const float* wi = w_ih + (size_t)(j * 256 + t) * 256;
    const float* wh = w_hh + (size_t)(j * 256 + t) * 256;
    float si = b_ih[j * 256 + t], sh = b_hh[j * 256 + t];
#pragma unroll 4
    for (int c = 0; c < 256; ++c) { si += xr[c] * wi[c]; sh += hr[c] * wh[c]; }
    gi[j] = si; gh[j] = sh;
  }
  const float rg = sigmoidf(gi[0] + gh[0]);
  const float zg = sigmoidf(gi[1] + gh[1]);
  const float ng = tanhf(gi[2] + rg * gh[2]);
  hn[t] = (1.0f - zg) * ng + zg * hr[t];
  __syncthreads();
  if (t < 26) {
    const float* wr = rw + (size_t)t * 256;
    float s = rb[t];
#pragma unroll 4
    for (int c = 0; c < 256; ++c) s += hn[c] * wr[c];
    out_labels[(size_t)bn * 26 + t] = s;
  } else if (t < 29) {
    const int tt = t - 26;
    const float* wr = rw2 + (size_t)tt * 256;
    float s = rb2[tt];
#pragma unroll 4
    for (int c = 0; c < 256; ++c) s += hn[c] * wr[c];
    out_roles[(size_t)bn * 3 + tt] = s;
  }
}

// ---------------------------------------------------------------------------
// Orchestration
// ---------------------------------------------------------------------------
extern "C" void kernel_launch(void* const* d_in, const int* in_sizes, int n_in,
                              void* d_out, int out_size, void* d_ws, size_t ws_size,
                              hipStream_t stream) {
  const float* ef   = (const float*)d_in[0];   // (B,N,N,CE)
  const float* nf   = (const float*)d_in[1];   // (B,N,CN)
  const float* lw1  = (const float*)d_in[4];
  const float* lb1  = (const float*)d_in[5];
  const float* lw2  = (const float*)d_in[6];
  const float* lb2  = (const float*)d_in[7];
  const float* lw3  = (const float*)d_in[8];   // (1,256)
  const float* lb3  = (const float*)d_in[9];   // (1,)
  const float* mw   = (const float*)d_in[10];  // (256,512)
  const float* mb   = (const float*)d_in[11];
  const float* w_ih = (const float*)d_in[12];
  const float* w_hh = (const float*)d_in[13];
  const float* b_ih = (const float*)d_in[14];
  const float* b_hh = (const float*)d_in[15];
  const float* rw   = (const float*)d_in[16];
  const float* rb   = (const float*)d_in[17];
  const float* rw2  = (const float*)d_in[18];
  const float* rb2  = (const float*)d_in[19];

  char* ws = (char*)d_ws;
  size_t off = 0;
  auto alloc = [&](size_t bytes) { char* p = ws + off; off += (bytes + 255) & ~size_t(255); return p; };
  unsigned short* E0b   = (unsigned short*)alloc((size_t)NELEM * 2);   // bf16 edge feats
  unsigned short* baseb = (unsigned short*)alloc((size_t)NELEM * 2);   // bf16 base
  unsigned short* Gb    = (unsigned short*)alloc((size_t)NELEM * 2);   // bf16 G = base@lw1^T
  unsigned short* H1b   = (unsigned short*)alloc((size_t)NELEM * 2);   // bf16 layer1 out
  unsigned short* lw1b  = (unsigned short*)alloc(65536 * 2);
  unsigned short* lw2b  = (unsigned short*)alloc(65536 * 2);
  unsigned short* mweb  = (unsigned short*)alloc(65536 * 2);
  float* hb    = (float*)alloc((size_t)BN * 256 * 4);
  float* Sbuf  = (float*)alloc((size_t)RWS * 4);
  float* Atmp  = (float*)alloc((size_t)RWS * 4);
  float* xbuf  = (float*)alloc((size_t)BN * 256 * 4);

  float* Aout       = (float*)d_out;            // [RWS]
  float* out_labels = Aout + RWS;               // [BN*26]
  float* out_roles  = out_labels + BN * 26;     // [BN*3]

  // 1) bf16 conversions
  f2bf_kernel<<<NELEM / 256, 256, 0, stream>>>(ef, E0b, NELEM);
  f2bf_kernel<<<65536 / 256, 256, 0, stream>>>(lw1, lw1b, 65536);
  f2bf_kernel<<<65536 / 256, 256, 0, stream>>>(lw2, lw2b, 65536);
  f2bf_strided_kernel<<<65536 / 256, 256, 0, stream>>>(mw, mweb, 256, 512, 0);

  // 2) node contribution to base
  hbase_kernel<<<BN, 256, 0, stream>>>(nf, mw, hb);

  // 3) base = relu(E0 @ mw_e^T + hb + mb)           (17.2 GFLOP)
  gemm_bf16_wmma<MODE_BASE><<<RWS / 128, 256, 0, stream>>>(
      E0b, mweb, baseb, mb, hb, nullptr, nullptr, nullptr, nullptr);

  // 4) G = base @ lw1^T  (reused by iterations 2,3) (17.2 GFLOP)
  gemm_bf16_wmma<MODE_G><<<RWS / 128, 256, 0, stream>>>(
      baseb, lw1b, Gb, nullptr, nullptr, nullptr, nullptr, nullptr, nullptr);

  // 5) iteration 1: H1 = relu(E0 @ lw1^T + lb1); A,S = layer2+layer3
  gemm_bf16_wmma<MODE_L1><<<RWS / 128, 256, 0, stream>>>(
      E0b, lw1b, H1b, lb1, nullptr, nullptr, nullptr, nullptr, nullptr);
  gemm_bf16_wmma<MODE_L2><<<RWS / 128, 256, 0, stream>>>(
      H1b, lw2b, nullptr, lb2, nullptr, lw3, lb3, Atmp, Sbuf);

  // 6) iteration 2: H1 = relu(S1 * G + lb1) (transposed rows), then layer2/3
  scale_relu_kernel<<<RWS, 256, 0, stream>>>(Gb, Sbuf, lb1, H1b);
  gemm_bf16_wmma<MODE_L2><<<RWS / 128, 256, 0, stream>>>(
      H1b, lw2b, nullptr, lb2, nullptr, lw3, lb3, Atmp, Sbuf);

  // 7) iteration 3: final A written straight into d_out
  scale_relu_kernel<<<RWS, 256, 0, stream>>>(Gb, Sbuf, lb1, H1b);
  gemm_bf16_wmma<MODE_L2><<<RWS / 128, 256, 0, stream>>>(
      H1b, lw2b, nullptr, lb2, nullptr, lw3, lb3, Aout, Sbuf);

  // 8) x[b,v,:] = sum_w S3[b,v,w] * base[(b,v,w),:]
  xsum_kernel<<<BN, 256, 0, stream>>>(Sbuf, baseb, xbuf);

  // 9) GRU + prediction heads
  gru_head_kernel<<<BN, 256, 0, stream>>>(xbuf, nf, w_ih, w_hh, b_ih, b_hh,
                                          rw, rb, rw2, rb2, out_labels, out_roles);
}
